// ModifiedGRUCell_39178691674857
// MI455X (gfx1250) — matrix operand order repaired and checked by reading, not compiled, not run on hardware
//
#include <hip/hip_runtime.h>
#include <hip/hip_bf16.h>

#define NB 16384

typedef __bf16 bf16_t;
typedef __attribute__((ext_vector_type(16))) __bf16 v16bf;
typedef __attribute__((ext_vector_type(8)))  __bf16 bf16x8;
typedef __attribute__((ext_vector_type(8)))  float  v8f;
typedef __attribute__((ext_vector_type(4)))  float  f32x4;

// ---------------------------------------------------------------------------
// Async global->LDS 16B copy (CDNA5, ASYNCcnt-tracked, no VGPR round trip).
// GVS addressing: mem = SADDR(64b) + VADDR(32b offset). First operand is the
// per-lane LDS byte address.
// ---------------------------------------------------------------------------
__device__ __forceinline__ void async_cp16(unsigned lds_byte,
                                           unsigned long long gbase,
                                           unsigned byte_off) {
  asm volatile("global_load_async_to_lds_b128 %0, %1, %2"
               :: "v"(lds_byte), "v"(byte_off), "s"(gbase)
               : "memory");
}
__device__ __forceinline__ unsigned lds_off(const void* p) {
  // LDS aperture occupies addr[63:32]; low 32 bits are the LDS byte offset.
  return (unsigned)(unsigned long long)(uintptr_t)p;
}

// ---------------------------------------------------------------------------
// f32 (strided rows) -> bf16 (dense 1024-wide), 4 elems/thread
// ---------------------------------------------------------------------------
__global__ __launch_bounds__(256)
void convert_bf16(const float* __restrict__ src, int ld, bf16_t* __restrict__ dst) {
  const size_t i = ((size_t)blockIdx.x * 256 + threadIdx.x) * 4;
  const int row = (int)(i >> 10);
  const int col = (int)(i & 1023);
  f32x4 v = *(const f32x4*)(src + (size_t)row * ld + col);
  bf16_t o[4] = {(bf16_t)v.x, (bf16_t)v.y, (bf16_t)v.z, (bf16_t)v.w};
  *(unsigned long long*)(dst + i) = *(unsigned long long*)o;
}

// ---------------------------------------------------------------------------
// Weight transpose + convert: W[K][1024] f32 -> Wt[1024][K] bf16
// ---------------------------------------------------------------------------
__global__ __launch_bounds__(256)
void transpose_convert(const float* __restrict__ W, bf16_t* __restrict__ Wt, int K) {
  __shared__ float t[32][33];
  const int tx = threadIdx.x & 31;
  const int ty = threadIdx.x >> 5;
  const int n0 = blockIdx.x * 32;
  const int k0 = blockIdx.y * 32;
#pragma unroll
  for (int j = 0; j < 32; j += 8)
    t[ty + j][tx] = W[(size_t)(k0 + ty + j) * 1024 + n0 + tx];
  __syncthreads();
#pragma unroll
  for (int j = 0; j < 32; j += 8)
    Wt[(size_t)(n0 + ty + j) * K + k0 + tx] = (bf16_t)t[tx][ty + j];
}

// ---------------------------------------------------------------------------
// bf16 WMMA GEMM, async double-buffered LDS, fused epilogue.
// out = act( concatK(A1,A2) @ Wt^T + bias ) (+ extra elementwise per MODE)
// BM=128, BN=128, BK=32; 8 waves; wave = 4(M) x 2(N) wmma tiles.
// ---------------------------------------------------------------------------
enum { MODE_RS = 0, MODE_U, MODE_C, MODE_RELU, MODE_FINAL };

template <int MODE>
__global__ __launch_bounds__(256)
void gemm_act(const bf16_t* __restrict__ A1,
              const bf16_t* __restrict__ A2,
              const bf16_t* __restrict__ Wt, int Ktot,
              const float* __restrict__ bias,
              const float*  __restrict__ e0,   // state (RS, FINAL)
              const bf16_t* __restrict__ e1,   // u     (FINAL)
              const bf16_t* __restrict__ e2,   // c     (FINAL)
              void* __restrict__ outv) {
  __shared__ bf16_t As[2][128][40];   // [buf][m][k] padded to 40 halfs
  __shared__ bf16_t Bs[2][128][40];   // [buf][n][k]

  const int tid  = threadIdx.x;
  const int lane = tid & 31;
  const int wave = tid >> 5;
  const int wm = wave >> 2;                 // 0..1
  const int wn = wave & 3;                  // 0..3
  const int blockN = blockIdx.x * 128;
  const int blockM = blockIdx.y * 128;

  v8f acc[4][2];
#pragma unroll
  for (int i = 0; i < 4; ++i)
#pragma unroll
    for (int j = 0; j < 2; ++j)
      acc[i][j] = v8f{0.f, 0.f, 0.f, 0.f, 0.f, 0.f, 0.f, 0.f};

  // staging mapping: each thread copies 2x16B of A and 2x16B of B per K-step
  const int srow  = tid >> 1;               // 0..127
  const int sbyte = (tid & 1) * 32;         // 0 or 32 within the 64B k-row

  const unsigned aRowByte = (unsigned)((blockM + srow) * 1024) * 2u;
  const unsigned bRowByte = (unsigned)((blockN + srow) * Ktot) * 2u;

  auto issue = [&](int buf, int k0) {
    const bf16_t* abase = (k0 < 1024) ? A1 : A2;
    const unsigned kk = (unsigned)((k0 < 1024) ? k0 : (k0 - 1024));
    const unsigned aOff = aRowByte + kk * 2u + (unsigned)sbyte;
    const unsigned aLds = lds_off(&As[buf][srow][0]) + (unsigned)sbyte;
    async_cp16(aLds,      (unsigned long long)(uintptr_t)abase, aOff);
    async_cp16(aLds + 16, (unsigned long long)(uintptr_t)abase, aOff + 16);
    const unsigned bOff = bRowByte + (unsigned)k0 * 2u + (unsigned)sbyte;
    const unsigned bLds = lds_off(&Bs[buf][srow][0]) + (unsigned)sbyte;
    async_cp16(bLds,      (unsigned long long)(uintptr_t)Wt, bOff);
    async_cp16(bLds + 16, (unsigned long long)(uintptr_t)Wt, bOff + 16);
  };

  const int mrow  = lane & 15;
  const int koffA = (lane >> 4) * 8;        // A frag K base per lane-half
  const int koffB = (lane >> 4) * 8;        // B frag K base per lane-half
  const int NK = Ktot >> 5;

  issue(0, 0);
  for (int kb = 0; kb < NK; ++kb) {
    const int cur = kb & 1;
    if (kb + 1 < NK) {
      issue(cur ^ 1, (kb + 1) << 5);
      // retire previous 4 async ops; keep the 4 just-issued in flight
      asm volatile("s_wait_asynccnt 0x4" ::: "memory");
    } else {
      asm volatile("s_wait_asynccnt 0x0" ::: "memory");
    }
    __syncthreads();

    v16bf afr[4], bfr[2];
#pragma unroll
    for (int mt = 0; mt < 4; ++mt) {
      const int m = wm * 64 + mt * 16 + mrow;
      ((bf16x8*)&afr[mt])[0] = *(const bf16x8*)&As[cur][m][koffA];
      ((bf16x8*)&afr[mt])[1] = *(const bf16x8*)&As[cur][m][koffA + 16];
    }
#pragma unroll
    for (int nt = 0; nt < 2; ++nt) {
      const int n = wn * 32 + nt * 16 + mrow;
      ((bf16x8*)&bfr[nt])[0] = *(const bf16x8*)&Bs[cur][n][koffB];
      ((bf16x8*)&bfr[nt])[1] = *(const bf16x8*)&Bs[cur][n][koffB + 16];
    }
#pragma unroll
    for (int mt = 0; mt < 4; ++mt)
#pragma unroll
      for (int nt = 0; nt < 2; ++nt)
        acc[mt][nt] = __builtin_amdgcn_wmma_f32_16x16x32_bf16(
            false, afr[mt], false, bfr[nt], (short)0, acc[mt][nt], false, false);
    __syncthreads();   // all reads done before this buffer is overwritten
  }

  // ---- epilogue ----
  const int rhalf = (lane >> 4) * 8;        // C/D: M = e + 8*(lane>>4), N = lane&15
#pragma unroll
  for (int nt = 0; nt < 2; ++nt) {
    const int gcol = blockN + wn * 32 + nt * 16 + mrow;
    const float bv = bias[gcol];
#pragma unroll
    for (int mt = 0; mt < 4; ++mt) {
#pragma unroll
      for (int e = 0; e < 8; ++e) {
        const int grow = blockM + wm * 64 + mt * 16 + rhalf + e;
        const size_t idx = (size_t)grow * 1024 + gcol;
        const float v = acc[mt][nt][e] + bv;
        if constexpr (MODE == MODE_RS) {
          const float r = 1.f / (1.f + __expf(-v));
          ((bf16_t*)outv)[idx] = (bf16_t)(r * e0[idx]);
        } else if constexpr (MODE == MODE_U) {
          ((bf16_t*)outv)[idx] = (bf16_t)(1.f / (1.f + __expf(-v)));
        } else if constexpr (MODE == MODE_C) {
          ((bf16_t*)outv)[idx] = (bf16_t)tanhf(v);
        } else if constexpr (MODE == MODE_RELU) {
          ((bf16_t*)outv)[idx] = (bf16_t)fmaxf(v, 0.f);
        } else {  // MODE_FINAL: new_h = u*s + (1-u)*c*g
          const float g = 1.f / (1.f + __expf(-v));
          const float u = (float)e1[idx];
          const float c = (float)e2[idx];
          const float s = e0[idx];
          ((float*)outv)[idx] = u * s + (1.f - u) * c * g;
        }
      }
    }
  }
}

// ---------------------------------------------------------------------------
extern "C" void kernel_launch(void* const* d_in, const int* in_sizes, int n_in,
                              void* d_out, int out_size, void* d_ws, size_t ws_size,
                              hipStream_t stream) {
  (void)in_sizes; (void)n_in; (void)out_size; (void)ws_size;
  const float* inputs = (const float*)d_in[0];
  const float* state  = (const float*)d_in[1];
  const float* Wr = (const float*)d_in[2];
  const float* br = (const float*)d_in[3];
  const float* Wu = (const float*)d_in[4];
  const float* bu = (const float*)d_in[5];
  const float* Wc = (const float*)d_in[6];
  const float* bc = (const float*)d_in[7];
  const float* W1 = (const float*)d_in[8];
  const float* b1 = (const float*)d_in[9];
  const float* W2 = (const float*)d_in[10];
  const float* b2 = (const float*)d_in[11];
  const float* Wg = (const float*)d_in[12];
  const float* bg = (const float*)d_in[13];
  float* out = (float*)d_out;

  char* ws = (char*)d_ws;
  size_t off = 0;
  auto alloc = [&](size_t bytes) -> void* {
    void* p = ws + off;
    off += (bytes + 255) & ~(size_t)255;
    return p;
  };

  bf16_t* WtR = (bf16_t*)alloc((size_t)2048 * 1024 * 2);
  bf16_t* WtU = (bf16_t*)alloc((size_t)2048 * 1024 * 2);
  bf16_t* WtC = (bf16_t*)alloc((size_t)2048 * 1024 * 2);
  bf16_t* WtG = (bf16_t*)alloc((size_t)2048 * 1024 * 2);
  bf16_t* Wt1 = (bf16_t*)alloc((size_t)1024 * 1024 * 2);
  bf16_t* Wt2 = (bf16_t*)alloc((size_t)1024 * 1024 * 2);
  bf16_t* xb  = (bf16_t*)alloc((size_t)NB * 1024 * 2);  // x (bf16)
  bf16_t* nbb = (bf16_t*)alloc((size_t)NB * 1024 * 2);  // neighbors (bf16)
  bf16_t* sb  = (bf16_t*)alloc((size_t)NB * 1024 * 2);  // state (bf16)
  bf16_t* uB  = (bf16_t*)alloc((size_t)NB * 1024 * 2);  // u
  bf16_t* cB  = (bf16_t*)alloc((size_t)NB * 1024 * 2);  // c
  bf16_t* rsB = (bf16_t*)alloc((size_t)NB * 1024 * 2);  // rs, reused as semb
  bf16_t* neB = (bf16_t*)alloc((size_t)NB * 1024 * 2);  // nemb

  const dim3 tb(256);
  const dim3 gC(NB * 1024 / (4 * 256));
  convert_bf16<<<gC, tb, 0, stream>>>(inputs,        2048, xb);
  convert_bf16<<<gC, tb, 0, stream>>>(inputs + 1024, 2048, nbb);
  convert_bf16<<<gC, tb, 0, stream>>>(state,         1024, sb);

  const dim3 gT2(32, 64), gT1(32, 32);
  transpose_convert<<<gT2, tb, 0, stream>>>(Wr, WtR, 2048);
  transpose_convert<<<gT2, tb, 0, stream>>>(Wu, WtU, 2048);
  transpose_convert<<<gT2, tb, 0, stream>>>(Wc, WtC, 2048);
  transpose_convert<<<gT2, tb, 0, stream>>>(Wg, WtG, 2048);
  transpose_convert<<<gT1, tb, 0, stream>>>(W1, Wt1, 1024);
  transpose_convert<<<gT1, tb, 0, stream>>>(W2, Wt2, 1024);

  const dim3 gG(1024 / 128, NB / 128);   // (8, 128)

  // rs = sigmoid([x||state]@Wr + br) * state
  gemm_act<MODE_RS><<<gG, tb, 0, stream>>>(xb, sb, WtR, 2048, br, state, nullptr, nullptr, rsB);
  // u = sigmoid([x||state]@Wu + bu)
  gemm_act<MODE_U><<<gG, tb, 0, stream>>>(xb, sb, WtU, 2048, bu, nullptr, nullptr, nullptr, uB);
  // c = tanh([x||rs]@Wc + bc)
  gemm_act<MODE_C><<<gG, tb, 0, stream>>>(xb, rsB, WtC, 2048, bc, nullptr, nullptr, nullptr, cB);
  // semb = relu(x@W1 + b1)   (reuses rsB after c is done)
  gemm_act<MODE_RELU><<<gG, tb, 0, stream>>>(xb, xb, Wt1, 1024, b1, nullptr, nullptr, nullptr, rsB);
  // nemb = relu(neighbors@W2 + b2)
  gemm_act<MODE_RELU><<<gG, tb, 0, stream>>>(nbb, nbb, Wt2, 1024, b2, nullptr, nullptr, nullptr, neB);
  // new_h = u*state + (1-u)*c*sigmoid([semb||nemb]@Wg + bg)
  gemm_act<MODE_FINAL><<<gG, tb, 0, stream>>>(rsB, neB, WtG, 2048, bg, state, uB, cB, out);
}
